// SparseMemoryBank_60352880443928
// MI455X (gfx1250) — matrix operand order repaired and checked by reading, not compiled, not run on hardware
//
#include <hip/hip_runtime.h>
#include <hip/hip_bf16.h>
#include <stdint.h>

#define NUM_SLOTS 128
#define HIDDEN    4096
#define T_WRITE   128
#define T_READ    8192
#define K_TOP     8
#define EMA_ALPHA 0.1f

typedef __attribute__((ext_vector_type(16))) __bf16 v16bf;
typedef __attribute__((ext_vector_type(8)))  float  v8f;

union BFPack {
    uint32_t u[8];
    v16bf    v;
};
union BF4Pack {
    __bf16 h[4];
    uint2  p;
};

// hardware float->bf16 (clang lowers to v_cvt_(pk_)bf16_f32 on gfx1250)
static __device__ __forceinline__ void cvt4(v16bf& v, int base, float4 f) {
    v[base + 0] = (__bf16)f.x;
    v[base + 1] = (__bf16)f.y;
    v[base + 2] = (__bf16)f.z;
    v[base + 3] = (__bf16)f.w;
}

// ---------------------------------------------------------------------------
// Kernel 1: gate + EMA scatter. slot_ids is a permutation, so mem_new is
// exactly {0.1*g[t]*values[t] + 0.9*mem_l[slot]} for every slot.
// Writes mem_new as fp32 (for epilogue gather) and bf16 (for WMMA B matrix).
// ---------------------------------------------------------------------------
__global__ void smb_write_kernel(const float* __restrict__ values,
                                 const float* __restrict__ memory,
                                 const float* __restrict__ gate_w,
                                 const float* __restrict__ gate_b,
                                 const int*   __restrict__ layer_idx,
                                 const int*   __restrict__ ptr,
                                 float*       __restrict__ memf,
                                 uint32_t*    __restrict__ membf /* 2 bf16 per u32 */) {
    __shared__ float red[256];
    const int t   = blockIdx.x;          // token 0..127
    const int tid = threadIdx.x;         // 0..255
    const int L   = layer_idx[0];
    const int P   = ptr[0];

    const float* vrow = values + (size_t)t * HIDDEN;
    const float* gw   = gate_w + (size_t)L * HIDDEN;

    // dot(values[t], gate_w[L])
    float partial = 0.0f;
    for (int d = tid; d < HIDDEN; d += 256) partial += vrow[d] * gw[d];
    red[tid] = partial;
    __syncthreads();
    for (int s = 128; s > 0; s >>= 1) {
        if (tid < s) red[tid] += red[tid + s];
        __syncthreads();
    }
    const float g = 1.0f / (1.0f + __expf(-(red[0] + gate_b[L])));

    const int slot = (P + t) & (NUM_SLOTS - 1);
    const float4* vrow4 = (const float4*)vrow;
    const float4* cur4  =
        (const float4*)(memory + ((size_t)L * NUM_SLOTS + slot) * HIDDEN);
    float4* dstf4 = (float4*)(memf + (size_t)slot * HIDDEN);
    uint2*  dstb2 = (uint2*)(membf) + (size_t)slot * (HIDDEN / 4);

    const float a = EMA_ALPHA * g;
    const float b = 1.0f - EMA_ALPHA;
    for (int i = tid; i < HIDDEN / 4; i += 256) {
        float4 v = vrow4[i];
        float4 c = cur4[i];
        float4 u;
        u.x = a * v.x + b * c.x;
        u.y = a * v.y + b * c.y;
        u.z = a * v.z + b * c.z;
        u.w = a * v.w + b * c.w;
        dstf4[i] = u;
        BF4Pack pk;
        pk.h[0] = (__bf16)u.x;
        pk.h[1] = (__bf16)u.y;
        pk.h[2] = (__bf16)u.z;
        pk.h[3] = (__bf16)u.w;
        dstb2[i] = pk.p;
    }
}

// ---------------------------------------------------------------------------
// Kernel 2: scores (WMMA bf16) -> top-8 -> softmax -> weighted gather-sum.
// 128 blocks x 128 threads (4 waves). Each wave: 16 queries x 128 slots.
// ---------------------------------------------------------------------------
__global__ void __launch_bounds__(128)
smb_read_kernel(const float* __restrict__ queries,
                const float* __restrict__ memf,
                const uint4* __restrict__ membf4, /* bf16 mem, 512 uint4/row */
                float*       __restrict__ out) {
    __shared__ uint4 Bt[512];              // 128 slots x 32 K bf16 = 8 KB
    __shared__ float sc[4][16][130];       // per-wave score tiles (padded)
    __shared__ float topw[4][16][K_TOP];
    __shared__ int   topi[4][16][K_TOP];

    const int tid   = threadIdx.x;
    const int wave  = tid >> 5;
    const int lane  = tid & 31;
    const int lrow  = lane & 15;           // M/N sub-index
    const int khalf = lane >> 4;           // lane half selects K sub-range
    const int qbase = blockIdx.x * 64 + wave * 16;

    const float4* Q4 = (const float4*)queries;

    v8f acc[8];
#pragma unroll
    for (int nt = 0; nt < 8; ++nt) acc[nt] = (v8f)0.0f;

    const int qr   = qbase + lrow;               // this lane's query row (A)
    const int qrow = qr * (HIDDEN / 4);          // float4 row base

    for (int kc = 0; kc < HIDDEN / 32; ++kc) {
        __syncthreads();  // prior iteration done reading Bt
        // ---- stage B chunk: membf[slot][kc*32 .. kc*32+31] -> LDS ----
#pragma unroll
        for (int j = 0; j < 4; ++j) {
            int u    = tid + 128 * j;            // 0..511
            int slot = u >> 2;
            int part = u & 3;
            Bt[u] = membf4[(size_t)slot * 512 + kc * 4 + part];
        }
        __syncthreads();

        // ---- A fragment: 16 fp32 -> bf16 per ISA 16-bit A layout ----
        // lanes 0-15: V0..3 = K 0..7, V4..7 = K 16..23 (of this 32-chunk)
        // lanes16-31: V0..3 = K 8..15, V4..7 = K 24..31
        const int qoff = qrow + kc * 8 + 2 * khalf;
        float4 f0 = Q4[qoff];
        float4 f1 = Q4[qoff + 1];
        float4 f2 = Q4[qoff + 4];
        float4 f3 = Q4[qoff + 5];
        v16bf av;
        cvt4(av, 0,  f0);
        cvt4(av, 4,  f1);
        cvt4(av, 8,  f2);
        cvt4(av, 12, f3);

        // ---- 8 slot-tiles of 16: B from LDS, WMMA accumulate ----
#pragma unroll
        for (int nt = 0; nt < 8; ++nt) {
            const int slot = nt * 16 + lrow;     // this lane's B column
            // lanes 0-15: K 0..7, 16..23 of chunk; lanes 16-31: K 8..15, 24..31
            uint4 b0 = Bt[slot * 4 + khalf];
            uint4 b1 = Bt[slot * 4 + khalf + 2];
            BFPack B;
            B.u[0] = b0.x; B.u[1] = b0.y; B.u[2] = b0.z; B.u[3] = b0.w;
            B.u[4] = b1.x; B.u[5] = b1.y; B.u[6] = b1.z; B.u[7] = b1.w;
#if defined(__HIP_DEVICE_COMPILE__)
            acc[nt] = __builtin_amdgcn_wmma_f32_16x16x32_bf16(
                false, av, false, B.v, (short)0, acc[nt], false, false);
#endif
        }
    }

    // ---- scatter scaled scores to LDS (C layout: VGPR r, lane half -> M) ----
    const float scale = 0.015625f;  // 1/sqrt(4096)
#pragma unroll
    for (int nt = 0; nt < 8; ++nt) {
#pragma unroll
        for (int r = 0; r < 8; ++r) {
            sc[wave][r + 8 * khalf][nt * 16 + lrow] = acc[nt][r] * scale;
        }
    }
    __syncthreads();

    // ---- per-query top-8 + softmax (lanes 0..15, register shift-insert) ----
    if (lane < 16) {
        float tv[K_TOP];
        int   ti8[K_TOP];
#pragma unroll
        for (int j = 0; j < K_TOP; ++j) { tv[j] = -3.4e38f; ti8[j] = 0; }
        const float* srow = &sc[wave][lane][0];
        for (int s = 0; s < NUM_SLOTS; ++s) {
            float cv = srow[s];
            int   ci = s;
#pragma unroll
            for (int j = 0; j < K_TOP; ++j) {   // keep sorted descending
                bool  p  = cv > tv[j];
                float nv = p ? cv : tv[j];
                int   ni = p ? ci : ti8[j];
                float ov = p ? tv[j] : cv;
                int   oi = p ? ti8[j] : ci;
                tv[j] = nv; ti8[j] = ni; cv = ov; ci = oi;
            }
        }
        float mx = tv[0];
        float e[K_TOP];
        float sum = 0.0f;
#pragma unroll
        for (int j = 0; j < K_TOP; ++j) { e[j] = __expf(tv[j] - mx); sum += e[j]; }
        float inv = 1.0f / sum;
#pragma unroll
        for (int j = 0; j < K_TOP; ++j) {
            topw[wave][lane][j] = e[j] * inv;
            topi[wave][lane][j] = ti8[j];
        }
    }
    __syncthreads();

    // ---- epilogue: out[q] = sum_j attn_j * mem_new[idx_j]  (fp32, L2-hot) ----
    const float4* M4 = (const float4*)memf;
    float4* O4 = (float4*)out;
    for (int q = 0; q < 16; ++q) {
        float w[K_TOP];
        const float4* rows[K_TOP];
#pragma unroll
        for (int j = 0; j < K_TOP; ++j) {
            w[j]    = topw[wave][q][j];
            rows[j] = M4 + (size_t)topi[wave][q][j] * (HIDDEN / 4);
        }
        const int obase = (qbase + q) * (HIDDEN / 4);
        for (int i = lane; i < HIDDEN / 4; i += 32) {
            float4 s = {0.0f, 0.0f, 0.0f, 0.0f};
#pragma unroll
            for (int j = 0; j < K_TOP; ++j) {
                float4 m = rows[j][i];
                s.x += w[j] * m.x;
                s.y += w[j] * m.y;
                s.z += w[j] * m.z;
                s.w += w[j] * m.w;
            }
            O4[obase + i] = s;
        }
    }
}

// ---------------------------------------------------------------------------
extern "C" void kernel_launch(void* const* d_in, const int* in_sizes, int n_in,
                              void* d_out, int out_size, void* d_ws, size_t ws_size,
                              hipStream_t stream) {
    const float* values  = (const float*)d_in[0];  // [128,4096]
    const float* queries = (const float*)d_in[1];  // [8192,4096]
    const float* memory  = (const float*)d_in[2];  // [32,128,4096]
    const float* gate_w  = (const float*)d_in[3];  // [32,4096]
    const float* gate_b  = (const float*)d_in[4];  // [32]
    const int*   layer   = (const int*)d_in[5];    // scalar
    // d_in[6] = k (hardcoded 8), d_in[7] = ptr
    const int*   ptr     = (const int*)d_in[7];
    float* out = (float*)d_out;                    // [8192,4096]

    // workspace: fp32 mem_new (2 MB) then bf16 mem_new (1 MB)
    float*    memf  = (float*)d_ws;
    uint32_t* membf = (uint32_t*)((char*)d_ws + (size_t)NUM_SLOTS * HIDDEN * 4);

    smb_write_kernel<<<T_WRITE, 256, 0, stream>>>(values, memory, gate_w, gate_b,
                                                  layer, ptr, memf, membf);
    smb_read_kernel<<<T_READ / 64, 128, 0, stream>>>(queries, memf,
                                                     (const uint4*)membf, out);
    (void)in_sizes; (void)n_in; (void)out_size; (void)ws_size;
}
